// SparseRouter_75058848464992
// MI455X (gfx1250) — matrix operand order repaired and checked by reading, not compiled, not run on hardware
//
#include <hip/hip_runtime.h>
#include <math.h>

// ---------------------------------------------------------------------------
// Problem constants (reference: B=8, N=2048, D=2048, E=64, CAP=64)
// ---------------------------------------------------------------------------
#define BB   8
#define NN   2048
#define DD   2048
#define EE   64
#define CAP  64
#define MM   (BB * NN)          // 16384 tokens
#define OUTSZ ((size_t)BB * NN * EE * CAP)   // 67,108,864 per output tensor

typedef __attribute__((ext_vector_type(16))) __bf16 v16bf;
typedef __attribute__((ext_vector_type(8)))  float  v8f;

// ---------------------------------------------------------------------------
// bf16 helpers (RNE) — manual bit manipulation so no header dependencies
// ---------------------------------------------------------------------------
__device__ __forceinline__ __bf16 f32_to_bf16(float f) {
  union { float f; unsigned u; } v; v.f = f;
  unsigned r = v.u + 0x7FFFu + ((v.u >> 16) & 1u);   // round to nearest even
  unsigned short h = (unsigned short)(r >> 16);
  return __builtin_bit_cast(__bf16, h);
}
__device__ __forceinline__ float bf16_to_f32(__bf16 b) {
  unsigned short h = __builtin_bit_cast(unsigned short, b);
  union { unsigned u; float f; } v; v.u = ((unsigned)h) << 16;
  return v.f;
}

// ---------------------------------------------------------------------------
// Kernel 1: repack w_gate [D,E] fp32 -> WMMA-B-fragment-ready bf16 hi/lo.
// B-frag k order chosen here (same as A, not AMD's table order): lanes 0-15 hold K=0..15 (elem j <-> K=j),
// lanes 16-31 hold K=16..31 (elem j <-> K=16+j). Stored per (kc, etile, lane)
// as 16 contiguous bf16 (32 B) so the GEMM loads them with one b256 per lane.
// ---------------------------------------------------------------------------
__global__ void pack_w_kernel(const float* __restrict__ W,
                              __bf16* __restrict__ Bhi,
                              __bf16* __restrict__ Blo) {
  int t = blockIdx.x * 256 + threadIdx.x;       // 64*4*32*16 = 131072 total
  if (t >= (DD / 32) * (EE / 16) * 32 * 16) return;
  int j    = t & 15;
  int lane = (t >> 4) & 31;
  int et   = (t >> 9) & 3;
  int kc   = t >> 11;
  int half = lane >> 4;
  int ncol = lane & 15;
  int K    = kc * 32 + half * 16 + j;
  int Ecol = et * 16 + ncol;
  float w  = W[(size_t)K * EE + Ecol];
  __bf16 h = f32_to_bf16(w);
  Bhi[t] = h;
  Blo[t] = f32_to_bf16(w - bf16_to_f32(h));
}

// ---------------------------------------------------------------------------
// Kernel 2: logits = X @ W via v_wmma_f32_16x16x32_bf16 with hi/lo split
// (fp32-accurate, 3 WMMAs per fragment pair). 256 thr = 8 waves; each wave
// owns a 16(M)x64(E) strip with 4 accumulators; K loop = 2048/32 = 64.
// A-frag (16x32 bf16): lane<16 -> row rowBase+lane, K = kb+{0..15};
// lane>=16 -> row rowBase+lane-16, K = kb+{16..31} (one 64B chunk).
// ---------------------------------------------------------------------------
__global__ void gemm_kernel(const float* __restrict__ X,      // [MM, DD]
                            const __bf16* __restrict__ Bhi,   // frag-packed
                            const __bf16* __restrict__ Blo,
                            float* __restrict__ logits) {     // [MM, EE]
  const int lane    = threadIdx.x & 31;
  const int wave    = threadIdx.x >> 5;
  const int rowBase = (blockIdx.x * 8 + wave) * 16;
  const int half    = lane >> 4;
  const int l15     = lane & 15;
  const int arow    = rowBase + l15;

  v8f acc0 = {}, acc1 = {}, acc2 = {}, acc3 = {};

  for (int kc = 0; kc < DD / 32; ++kc) {
    // ---- A fragment: hi/lo bf16 from fp32 row-major tokens ----
    const float* xrow = X + (size_t)arow * DD + kc * 32 + half * 16;
    float4 x0 = *(const float4*)(xrow + 0);
    float4 x1 = *(const float4*)(xrow + 4);
    float4 x2 = *(const float4*)(xrow + 8);
    float4 x3 = *(const float4*)(xrow + 12);
    float xs[16] = { x0.x, x0.y, x0.z, x0.w, x1.x, x1.y, x1.z, x1.w,
                     x2.x, x2.y, x2.z, x2.w, x3.x, x3.y, x3.z, x3.w };
    v16bf ahi, alo;
#pragma unroll
    for (int j = 0; j < 16; ++j) {
      __bf16 h = f32_to_bf16(xs[j]);
      ahi[j] = h;
      alo[j] = f32_to_bf16(xs[j] - bf16_to_f32(h));
    }

    // ---- B fragments: contiguous 32B per lane, hot in WGP$/L2 ----
    const size_t fbase = (size_t)kc * 2048 + (size_t)lane * 16;
    v16bf bh0 = *(const v16bf*)(Bhi + fbase + 0 * 512);
    v16bf bl0 = *(const v16bf*)(Blo + fbase + 0 * 512);
    v16bf bh1 = *(const v16bf*)(Bhi + fbase + 1 * 512);
    v16bf bl1 = *(const v16bf*)(Blo + fbase + 1 * 512);
    v16bf bh2 = *(const v16bf*)(Bhi + fbase + 2 * 512);
    v16bf bl2 = *(const v16bf*)(Blo + fbase + 2 * 512);
    v16bf bh3 = *(const v16bf*)(Bhi + fbase + 3 * 512);
    v16bf bl3 = *(const v16bf*)(Blo + fbase + 3 * 512);

    // ---- 12 WMMAs: hi*hi + hi*lo + lo*hi per E-tile ----
    acc0 = __builtin_amdgcn_wmma_f32_16x16x32_bf16(false, ahi, false, bh0, (short)0, acc0, false, false);
    acc0 = __builtin_amdgcn_wmma_f32_16x16x32_bf16(false, ahi, false, bl0, (short)0, acc0, false, false);
    acc0 = __builtin_amdgcn_wmma_f32_16x16x32_bf16(false, alo, false, bh0, (short)0, acc0, false, false);
    acc1 = __builtin_amdgcn_wmma_f32_16x16x32_bf16(false, ahi, false, bh1, (short)0, acc1, false, false);
    acc1 = __builtin_amdgcn_wmma_f32_16x16x32_bf16(false, ahi, false, bl1, (short)0, acc1, false, false);
    acc1 = __builtin_amdgcn_wmma_f32_16x16x32_bf16(false, alo, false, bh1, (short)0, acc1, false, false);
    acc2 = __builtin_amdgcn_wmma_f32_16x16x32_bf16(false, ahi, false, bh2, (short)0, acc2, false, false);
    acc2 = __builtin_amdgcn_wmma_f32_16x16x32_bf16(false, ahi, false, bl2, (short)0, acc2, false, false);
    acc2 = __builtin_amdgcn_wmma_f32_16x16x32_bf16(false, alo, false, bh2, (short)0, acc2, false, false);
    acc3 = __builtin_amdgcn_wmma_f32_16x16x32_bf16(false, ahi, false, bh3, (short)0, acc3, false, false);
    acc3 = __builtin_amdgcn_wmma_f32_16x16x32_bf16(false, ahi, false, bl3, (short)0, acc3, false, false);
    acc3 = __builtin_amdgcn_wmma_f32_16x16x32_bf16(false, alo, false, bh3, (short)0, acc3, false, false);
  }

  // C/D layout: VGPR r -> rows (half?8+r:r), col = lane&15
#pragma unroll
  for (int r = 0; r < 8; ++r) {
    size_t row = (size_t)rowBase + half * 8 + r;
    float* lp = logits + row * EE + l15;
    lp[0]  = acc0[r];
    lp[16] = acc1[r];
    lp[32] = acc2[r];
    lp[48] = acc3[r];
  }
}

// ---------------------------------------------------------------------------
// Kernel 3: per-token max / first-argmax / logsumexp over E=64.
// One wave32 per token, 2 logits per lane, butterfly shuffles.
// ---------------------------------------------------------------------------
__global__ void router_stats_kernel(const float* __restrict__ logits,
                                    float* __restrict__ gate,
                                    int*   __restrict__ idx,
                                    float* __restrict__ z2) {
  const int lane  = threadIdx.x & 31;
  const int wave  = threadIdx.x >> 5;
  const int token = blockIdx.x * 8 + wave;
  const float* lp = logits + (size_t)token * EE;
  float2 v = *(const float2*)(lp + lane * 2);

  float m  = (v.y > v.x) ? v.y : v.x;
  int   mi = (v.y > v.x) ? (lane * 2 + 1) : (lane * 2);
#pragma unroll
  for (int off = 16; off > 0; off >>= 1) {
    float om = __shfl_xor(m, off, 32);
    int   oi = __shfl_xor(mi, off, 32);
    if (om > m || (om == m && oi < mi)) { m = om; mi = oi; }
  }
  float s = __expf(v.x - m) + __expf(v.y - m);
#pragma unroll
  for (int off = 16; off > 0; off >>= 1) s += __shfl_xor(s, off, 32);

  if (lane == 0) {
    float lse = m + __logf(s);
    gate[token] = 1.0f / s;       // == exp(max - lse) exactly
    idx[token]  = mi;
    z2[token]   = lse * lse;
  }
}

// ---------------------------------------------------------------------------
// Kernel 4: deterministic z-loss reduction (fixed tree, no float atomics).
// ---------------------------------------------------------------------------
__global__ void zloss_kernel(const float* __restrict__ z2,
                             float* __restrict__ out) {
  __shared__ float sm[256];
  float s = 0.0f;
  for (int i = threadIdx.x; i < MM; i += 256) s += z2[i];
  sm[threadIdx.x] = s;
  __syncthreads();
  for (int off = 128; off > 0; off >>= 1) {
    if ((int)threadIdx.x < off) sm[threadIdx.x] += sm[threadIdx.x + off];
    __syncthreads();
  }
  if (threadIdx.x == 0) *out = sm[0] / (float)MM;
}

// ---------------------------------------------------------------------------
// Kernel 5: position-in-expert via segmented prefix sum.
// One block per (batch, expert); LDS Hillis-Steele scan over 8 chunks of 256.
// Each token is written by exactly one (its chosen expert's) block.
// ---------------------------------------------------------------------------
__global__ void positions_kernel(const int* __restrict__ idx,
                                 int* __restrict__ pos) {
  __shared__ int sm[256];
  const int b = blockIdx.x >> 6;
  const int e = blockIdx.x & 63;
  const int* ib = idx + b * NN;
  int running = 0;
  for (int chunk = 0; chunk < NN / 256; ++chunk) {
    const int n = chunk * 256 + threadIdx.x;
    const int flag = (ib[n] == e) ? 1 : 0;
    sm[threadIdx.x] = flag;
    __syncthreads();
    for (int off = 1; off < 256; off <<= 1) {
      int add = 0;
      if ((int)threadIdx.x >= off) add = sm[threadIdx.x - off];
      __syncthreads();
      sm[threadIdx.x] += add;
      __syncthreads();
    }
    const int incl  = sm[threadIdx.x];
    const int total = sm[255];
    if (flag) pos[b * NN + n] = running + incl - 1;
    running += total;
    __syncthreads();
  }
}

// ---------------------------------------------------------------------------
// Kernel 6: fused one-hot materialization of dispatch_mask + combine_array.
// 537 MB written exactly once with float4 stores; idx/pos/gate (192 KB) stay
// resident in L2. Capacity drop (pos >= 64) falls out of the c-compare.
// ---------------------------------------------------------------------------
__global__ void write_outputs_kernel(const int*   __restrict__ idx,
                                     const int*   __restrict__ pos,
                                     const float* __restrict__ gate,
                                     float* __restrict__ dispatch,
                                     float* __restrict__ combine) {
  const size_t t    = (size_t)blockIdx.x * 256 + threadIdx.x;
  const size_t elem = t * 4;                 // index into [B,N,E,C]
  const int c     = (int)(elem & (CAP - 1));
  const int e     = (int)((elem >> 6) & (EE - 1));
  const size_t tok = elem >> 12;             // b*N + n  (E*C = 4096)

  const int   ii = idx[tok];
  const int   p  = pos[tok];
  const float g  = gate[tok];
  const bool  m  = (ii == e);

  float4 d, cm;
  d.x = (m && p == c + 0) ? 1.0f : 0.0f;
  d.y = (m && p == c + 1) ? 1.0f : 0.0f;
  d.z = (m && p == c + 2) ? 1.0f : 0.0f;
  d.w = (m && p == c + 3) ? 1.0f : 0.0f;
  cm.x = d.x * g; cm.y = d.y * g; cm.z = d.z * g; cm.w = d.w * g;

  *(float4*)(dispatch + elem) = d;
  *(float4*)(combine  + elem) = cm;
}

// ---------------------------------------------------------------------------
// Launch
// ---------------------------------------------------------------------------
extern "C" void kernel_launch(void* const* d_in, const int* in_sizes, int n_in,
                              void* d_out, int out_size, void* d_ws, size_t ws_size,
                              hipStream_t stream) {
  (void)in_sizes; (void)n_in; (void)out_size; (void)ws_size;
  const float* X = (const float*)d_in[0];    // [8,2048,2048]
  const float* W = (const float*)d_in[1];    // [2048,64]
  // d_in[2] = expert_capacity (==64, compile-time constant here)

  char* ws = (char*)d_ws;                    // ~4.75 MB used
  float*  logits = (float*) (ws + 0);                      // 4 MB
  __bf16* Bhi    = (__bf16*)(ws + 4 * 1024 * 1024);        // 256 KB
  __bf16* Blo    = (__bf16*)(ws + 4 * 1024 * 1024 + 262144);
  float*  gate   = (float*) (ws + 4 * 1024 * 1024 + 524288);
  int*    idx    = (int*)   (ws + 4 * 1024 * 1024 + 524288 + 65536);
  float*  z2     = (float*) (ws + 4 * 1024 * 1024 + 524288 + 131072);
  int*    pos    = (int*)   (ws + 4 * 1024 * 1024 + 524288 + 196608);

  float* dispatch = (float*)d_out;
  float* combine  = dispatch + OUTSZ;
  float* zloss    = dispatch + 2 * OUTSZ;

  pack_w_kernel<<<512, 256, 0, stream>>>(W, Bhi, Blo);
  gemm_kernel<<<MM / 128, 256, 0, stream>>>(X, Bhi, Blo, logits);
  router_stats_kernel<<<MM / 8, 256, 0, stream>>>(logits, gate, idx, z2);
  zloss_kernel<<<1, 256, 0, stream>>>(z2, zloss);
  positions_kernel<<<BB * EE, 256, 0, stream>>>(idx, pos);
  write_outputs_kernel<<<(unsigned)(OUTSZ / 4 / 256), 256, 0, stream>>>(
      idx, pos, gate, dispatch, combine);
}
